// SelectiveSSM_24644522344831
// MI455X (gfx1250) — matrix operand order repaired and checked by reading, not compiled, not run on hardware
//
#include <hip/hip_runtime.h>
#include <hip/hip_bf16.h>
#include <cstdint>
#include <cstddef>

// Problem dims (fixed by reference)
#define B_  2
#define L_  2048
#define DM_ 1024
#define DI_ 2048
#define DS_ 16
#define DC_ 4
#define DR_ 64

typedef __attribute__((ext_vector_type(16))) __bf16 v16bf;
typedef __attribute__((ext_vector_type(8)))  float  v8f;
typedef __attribute__((ext_vector_type(4)))  unsigned tdm_g0_t;
typedef __attribute__((ext_vector_type(8)))  unsigned tdm_g1_t;
typedef unsigned short u16t;
typedef unsigned int   u32t;

struct U128x2 { uint4 a, b; };

__device__ __forceinline__ u16t f2bf(float f) {
  u32t u = __builtin_bit_cast(u32t, f);
  u += 0x7FFFu + ((u >> 16) & 1u);          // round-to-nearest-even
  return (u16t)(u >> 16);
}

// ---------------------------------------------------------------------------
// gfx1250 async global->LDS copy (16 bytes per lane), tracked by ASYNCcnt
// ---------------------------------------------------------------------------
__device__ __forceinline__ void async_copy_b128(const u16t* gsrc, u16t* lds_dst) {
  unsigned lds_off = (unsigned)(uintptr_t)lds_dst;      // low 32b of flat = LDS byte offset
  unsigned long long ga = (unsigned long long)(uintptr_t)gsrc;
  asm volatile("global_load_async_to_lds_b128 %0, %1, off"
               :: "v"(lds_off), "v"(ga)
               : "memory");
}
__device__ __forceinline__ void wait_async0() {
  asm volatile("s_wait_asynccnt 0x0" ::: "memory");
}

// ---------------------------------------------------------------------------
// gfx1250 Tensor Data Mover: one 2D tile (tile_d0 cols x tile_d1 rows) of
// 2-byte elements, row stride `stride_el`, global -> LDS with hardware row
// padding: pad_interval=32 DWORDs (128B = 64 bf16), pad_amount=4 DWORDs (16B)
// => LDS row pitch 144B, matching the 72-element padded layout below.
// Tracked by TENSORcnt.
// ---------------------------------------------------------------------------
__device__ __forceinline__ void tdm_load_2d_b16(unsigned lds_off, const void* gsrc,
                                                unsigned tile_d0, unsigned tile_d1,
                                                unsigned stride_el) {
  unsigned long long ga = (unsigned long long)(uintptr_t)gsrc;
  const unsigned td0 = 0x7FFFFFFFu;   // tensor dims: large (no OOB clipping wanted)
  const unsigned td1 = 0x7FFFFFFFu;
  tdm_g0_t g0;
  g0[0] = 1u;                                          // count=1 (valid user descriptor)
  g0[1] = lds_off;                                     // lds_addr
  g0[2] = (unsigned)ga;                                // global_addr[31:0]  (bits 95:64)
  g0[3] = (unsigned)((ga >> 32) & 0x1FFFFFFu) | (2u << 30); // addr[56:32], type=2
  tdm_g1_t g1;
  g1[0] = (1u << 16)                                   // data_size = 2 bytes
        | (1u << 20)                                   // pad_enable
        | (4u << 22)                                   // pad_interval: 32 DWORDs
        | (3u << 25);                                  // pad_amount: 4 DWORDs
  g1[1] = (td0 & 0xFFFFu) << 16;                       // tensor_dim0[15:0]
  g1[2] = (td0 >> 16) | ((td1 & 0xFFFFu) << 16);       // tensor_dim0[31:16], dim1[15:0]
  g1[3] = (td1 >> 16) | (tile_d0 << 16);               // tensor_dim1[31:16], tile_dim0
  g1[4] = tile_d1;                                     // tile_dim1 (tile_dim2 = 0)
  g1[5] = stride_el;                                   // tensor_dim0_stride[31:0]
  g1[6] = 0u;                                          // stride[47:32], dim1_stride lo
  g1[7] = 0u;
  asm volatile("tensor_load_to_lds %0, %1"
               :: "s"(g0), "s"(g1)
               : "memory");
}

// ---------------------------------------------------------------------------
// fp32 -> bf16 conversion kernels
// ---------------------------------------------------------------------------
__global__ __launch_bounds__(256) void cvt_f32_bf16(const float* __restrict__ in,
                                                    u16t* __restrict__ out, int n) {
  int i = blockIdx.x * 256 + threadIdx.x;
  if (i < n) out[i] = f2bf(in[i]);
}

// row-major (rows_in x cols) -> (rows_out x cols), zero-padded tail rows
__global__ __launch_bounds__(256) void cvt_pad_rows(const float* __restrict__ in,
                                                    u16t* __restrict__ out,
                                                    int rows_in, int rows_out, int cols) {
  int i = blockIdx.x * 256 + threadIdx.x;
  if (i >= rows_out * cols) return;
  int r = i / cols;
  out[i] = (r < rows_in) ? f2bf(in[i]) : (u16t)0;
}

// extract dt slice (cols 0..63) from padded x_dbl (4096 x 128) as bf16 (4096 x 64)
__global__ __launch_bounds__(256) void slice_cvt_dtin(const float* __restrict__ xdbl,
                                                      u16t* __restrict__ out) {
  int i = blockIdx.x * 256 + threadIdx.x;   // over 4096*64
  int r = i >> 6, c = i & 63;
  out[i] = f2bf(xdbl[r * 128 + c]);
}

// ---------------------------------------------------------------------------
// bf16 WMMA GEMM:  C(M,N) = A(M,K) * W(N,K)^T   (+ optional softplus(x+bias))
// block tile 128x128, 8 waves (2x4), wave tile 64x32, K-step 64
// A tile: per-lane global_load_async_to_lds_b128 (ASYNCcnt)
// W tile: single TDM tensor_load_to_lds descriptor, wave 0 (TENSORcnt)
// ---------------------------------------------------------------------------
#define EPI_NONE 0
#define EPI_SOFTPLUS_BIAS 1

template<int EPI>
__global__ __launch_bounds__(256) void gemm_bf16_wmma(
    const u16t* __restrict__ A, const u16t* __restrict__ W,
    float* __restrict__ C, const float* __restrict__ bias,
    int M, int N, int K, int lda, int ldb, int ldc)
{
  __shared__ u16t As[2][128 * 72];  // 64 data cols + 8 pad, 144B row stride (16B aligned)
  __shared__ u16t Bs[2][128 * 72];

  const int tid  = threadIdx.x;
  const int lane = tid & 31;
  const int wave = tid >> 5;
  const int wm   = wave & 1;          // 2 waves along M
  const int wn   = wave >> 1;         // 4 waves along N
  const int h    = lane >> 4;         // lane half (0/1)
  const int mn   = lane & 15;
  const int m0   = blockIdx.y * 128;
  const int n0   = blockIdx.x * 128;

  v8f acc[4][2] = {};                 // 4 m-tiles x 2 n-tiles of 16x16

  // stage one 128x64 bf16 tile pair into LDS buffer `buf`
  auto stage = [&](int buf, int kb) {
    // A tile: 4 x 16B async chunks per thread
    #pragma unroll
    for (int i = 0; i < 4; ++i) {
      int cid = tid + i * 256;        // 0..1023
      int r = cid >> 3, c8 = cid & 7; // row 0..127, 16B-chunk 0..7
      async_copy_b128(A + (size_t)(m0 + r) * lda + kb + c8 * 8, &As[buf][r * 72 + c8 * 8]);
    }
    // W tile: one TDM 2D descriptor (64 cols x 128 rows), issued by wave 0 only
    if (wave == 0) {
      tdm_load_2d_b16((unsigned)(uintptr_t)&Bs[buf][0],
                      W + (size_t)n0 * ldb + kb,
                      64u, 128u, (unsigned)ldb);
    }
  };

  stage(0, 0);
  int ping = 0;

  for (int kb = 0; kb < K; kb += 64) {
    wait_async0();                          // this wave's async->LDS writes landed
    __builtin_amdgcn_s_wait_tensorcnt(0);   // wave 0's TDM transfer landed
    __syncthreads();                        // all writes visible; prev compute done
    if (kb + 64 < K) stage(ping ^ 1, kb + 64);  // overlap next tile with compute

    const u16t* Ab = &As[ping][0];
    const u16t* Bb = &Bs[ping][0];

    #pragma unroll
    for (int s = 0; s < 2; ++s) {
      const int k0 = s * 32;
      // preload ALL fragments for this K-32 step, then run 8 WMMAs back-to-back
      v16bf bf[2];
      #pragma unroll
      for (int j = 0; j < 2; ++j) {
        // B fragment (32x16): lane group h holds K = 16h..16h+15 for column mn
        const u16t* p = Bb + (wn * 32 + j * 16 + mn) * 72 + k0 + 16 * h;
        U128x2 raw{ *(const uint4*)p, *(const uint4*)(p + 8) };
        bf[j] = __builtin_bit_cast(v16bf, raw);
      }
      v16bf af[4];
      #pragma unroll
      for (int i = 0; i < 4; ++i) {
        // A fragment (16x32): lane group h holds K = 16h..16h+15 for row mn
        const u16t* p = Ab + (wm * 64 + i * 16 + mn) * 72 + k0 + 16 * h;
        U128x2 raw{ *(const uint4*)p, *(const uint4*)(p + 8) };
        af[i] = __builtin_bit_cast(v16bf, raw);
      }
      #pragma unroll
      for (int i = 0; i < 4; ++i) {
        acc[i][0] = __builtin_amdgcn_wmma_f32_16x16x32_bf16(
            false, af[i], false, bf[0], (short)0, acc[i][0], false, false);
        acc[i][1] = __builtin_amdgcn_wmma_f32_16x16x32_bf16(
            false, af[i], false, bf[1], (short)0, acc[i][1], false, false);
      }
    }
    ping ^= 1;
  }

  // ---- epilogue: D layout VGPR v -> row v + 8h, col = lane mn
  #pragma unroll
  for (int i = 0; i < 4; ++i) {
    #pragma unroll
    for (int j = 0; j < 2; ++j) {
      const int gcol  = n0 + wn * 32 + j * 16 + mn;
      const int grow0 = m0 + wm * 64 + i * 16 + 8 * h;
      #pragma unroll
      for (int v = 0; v < 8; ++v) {
        float val = acc[i][j][v];
        if (EPI == EPI_SOFTPLUS_BIAS) {
          val += bias[gcol];
          val = (val > 20.0f) ? val : log1pf(__expf(val));
        }
        C[(size_t)(grow0 + v) * ldc + gcol] = val;
      }
    }
  }
}

// ---------------------------------------------------------------------------
// causal depthwise conv1d (k=4) + SiLU; writes fp32 and bf16 copies
// ---------------------------------------------------------------------------
__global__ __launch_bounds__(256) void conv_silu_kernel(
    const float* __restrict__ xz, const float* __restrict__ cw,
    const float* __restrict__ cb, float* __restrict__ xf, u16t* __restrict__ xb)
{
  int i   = blockIdx.x * 256 + threadIdx.x;   // over B_*L_*DI_
  int d   = i & (DI_ - 1);
  int row = i >> 11;                           // DI_ = 2048
  int l   = row & (L_ - 1);
  int b   = row >> 11;                         // L_ = 2048
  float acc = cb[d];
  #pragma unroll
  for (int k = 0; k < DC_; ++k) {
    int ls = l - (DC_ - 1) + k;
    if (ls >= 0)
      acc += cw[d * DC_ + k] * xz[(size_t)(b * L_ + ls) * (2 * DI_) + d];
  }
  float sv = acc / (1.0f + __expf(-acc));      // SiLU
  xf[i] = sv;
  xb[i] = f2bf(sv);
}

// ---------------------------------------------------------------------------
// selective scan: lane = (channel-pair g, state n); 16-lane butterfly for y
// grid (DI_/16, B_), block 256 (8 waves x 2 channels x 16 states)
// ---------------------------------------------------------------------------
__global__ __launch_bounds__(256) void scan_kernel(
    const float* __restrict__ dt, const float* __restrict__ xdbl,
    const float* __restrict__ u, const float* __restrict__ A_log,
    float* __restrict__ y)
{
  const int b    = blockIdx.y;
  const int lane = threadIdx.x & 31;
  const int wave = threadIdx.x >> 5;
  const int g    = lane >> 4;                  // which channel of this wave's pair
  const int n    = lane & 15;                  // state index
  const int d    = blockIdx.x * 16 + wave * 2 + g;

  const float Ad = -__expf(A_log[d * DS_ + n]);
  float s = 0.0f;

  const float* dt_p = dt  + (size_t)b * L_ * DI_ + d;
  const float* u_p  = u   + (size_t)b * L_ * DI_ + d;
  const float* bd   = xdbl + (size_t)b * L_ * 128;
  float*       y_p  = y   + (size_t)b * L_ * DI_ + d;

  for (int l = 0; l < L_; ++l) {
    float dtv = dt_p[(size_t)l * DI_];
    float uv  = u_p[(size_t)l * DI_];
    float Bn  = bd[l * 128 + DR_ + n];
    float Cn  = bd[l * 128 + DR_ + DS_ + n];
    s = __expf(dtv * Ad) * s + dtv * Bn * uv;
    float p = s * Cn;
    p += __shfl_xor(p, 8, 32);
    p += __shfl_xor(p, 4, 32);
    p += __shfl_xor(p, 2, 32);
    p += __shfl_xor(p, 1, 32);
    if (n == 0) y_p[(size_t)l * DI_] = p;
  }
}

// ---------------------------------------------------------------------------
// gating: yg = (y + x*D) * silu(z), output bf16 for out_proj
// ---------------------------------------------------------------------------
__global__ __launch_bounds__(256) void gate_kernel(
    const float* __restrict__ y, const float* __restrict__ xf,
    const float* __restrict__ xz, const float* __restrict__ Dp,
    u16t* __restrict__ yg)
{
  int i   = blockIdx.x * 256 + threadIdx.x;   // over B_*L_*DI_
  int d   = i & (DI_ - 1);
  int row = i >> 11;
  float z   = xz[(size_t)row * (2 * DI_) + DI_ + d];
  float val = (y[i] + xf[i] * Dp[d]) * (z / (1.0f + __expf(-z)));
  yg[i] = f2bf(val);
}

// ---------------------------------------------------------------------------
// launcher
// ---------------------------------------------------------------------------
extern "C" void kernel_launch(void* const* d_in, const int* in_sizes, int n_in,
                              void* d_out, int out_size, void* d_ws, size_t ws_size,
                              hipStream_t stream) {
  const float* hs   = (const float*)d_in[0];  // (B,L,DM)
  const float* ipw  = (const float*)d_in[1];  // (2*DI, DM)
  const float* cw   = (const float*)d_in[2];  // (DI,1,DC)
  const float* cb   = (const float*)d_in[3];  // (DI)
  const float* xpw  = (const float*)d_in[4];  // (DR+2*DS, DI)
  const float* dtw  = (const float*)d_in[5];  // (DI, DR)
  const float* dtb  = (const float*)d_in[6];  // (DI)
  const float* Alog = (const float*)d_in[7];  // (DI, DS)
  const float* Dp   = (const float*)d_in[8];  // (DI)
  const float* opw  = (const float*)d_in[9];  // (DM, DI)

  const size_t M = (size_t)B_ * L_;           // 4096 token rows

  // workspace layout (256B aligned slabs)
  char* ws = (char*)d_ws;
  size_t off = 0;
  auto alloc = [&](size_t bytes) { size_t r = off; off += (bytes + 255) & ~(size_t)255; return r; };
  size_t o_hsb  = alloc(M * DM_ * 2);              // hidden bf16
  size_t o_ipw  = alloc((size_t)2 * DI_ * DM_ * 2);// in_proj_w bf16
  size_t o_xz   = alloc(M * 2 * DI_ * 4);          // xz fp32
  size_t o_xf   = alloc(M * DI_ * 4);              // conv(x) fp32
  size_t o_xb   = alloc(M * DI_ * 2);              // conv(x) bf16
  size_t o_xpw  = alloc((size_t)128 * DI_ * 2);    // x_proj_w bf16, padded 96->128
  size_t o_xdbl = alloc(M * 128 * 4);              // x_dbl fp32, padded ld=128
  size_t o_dtin = alloc(M * DR_ * 2);              // dt input bf16
  size_t o_dtw  = alloc((size_t)DI_ * DR_ * 2);    // dt_w bf16
  size_t o_dtf  = alloc(M * DI_ * 4);              // dt fp32 (post softplus)
  size_t o_yf   = alloc(M * DI_ * 4);              // scan output fp32
  size_t o_yg   = alloc(M * DI_ * 2);              // gated y bf16
  size_t o_opw  = alloc((size_t)DM_ * DI_ * 2);    // out_proj_w bf16
  (void)ws_size; (void)in_sizes; (void)n_in; (void)out_size;

  u16t*  hsb_p  = (u16t*)(ws + o_hsb);
  u16t*  ipw_p  = (u16t*)(ws + o_ipw);
  float* xz_p   = (float*)(ws + o_xz);
  float* xf_p   = (float*)(ws + o_xf);
  u16t*  xb_p   = (u16t*)(ws + o_xb);
  u16t*  xpw_p  = (u16t*)(ws + o_xpw);
  float* xdbl_p = (float*)(ws + o_xdbl);
  u16t*  dtin_p = (u16t*)(ws + o_dtin);
  u16t*  dtw_p  = (u16t*)(ws + o_dtw);
  float* dtf_p  = (float*)(ws + o_dtf);
  float* yf_p   = (float*)(ws + o_yf);
  u16t*  yg_p   = (u16t*)(ws + o_yg);
  u16t*  opw_p  = (u16t*)(ws + o_opw);

  // 1) conversions
  cvt_f32_bf16<<<(int)(M * DM_ / 256), 256, 0, stream>>>(hs, hsb_p, (int)(M * DM_));
  cvt_f32_bf16<<<(int)(2 * DI_ * DM_ / 256), 256, 0, stream>>>(ipw, ipw_p, 2 * DI_ * DM_);
  cvt_pad_rows<<<(128 * DI_ + 255) / 256, 256, 0, stream>>>(xpw, xpw_p, DR_ + 2 * DS_, 128, DI_);
  cvt_f32_bf16<<<(DI_ * DR_ / 256), 256, 0, stream>>>(dtw, dtw_p, DI_ * DR_);
  cvt_f32_bf16<<<(DM_ * DI_ / 256), 256, 0, stream>>>(opw, opw_p, DM_ * DI_);

  // 2) in_proj: xz(4096,4096) = hs(4096,1024) x in_proj_w(4096,1024)^T
  gemm_bf16_wmma<EPI_NONE><<<dim3(4096 / 128, (int)(M / 128)), 256, 0, stream>>>(
      hsb_p, ipw_p, xz_p, nullptr, (int)M, 2 * DI_, DM_, DM_, DM_, 2 * DI_);

  // 3) causal depthwise conv + SiLU
  conv_silu_kernel<<<(int)(M * DI_ / 256), 256, 0, stream>>>(xz_p, cw, cb, xf_p, xb_p);

  // 4) x_proj: x_dbl(4096,128pad) = x(4096,2048) x x_proj_w_pad(128,2048)^T
  gemm_bf16_wmma<EPI_NONE><<<dim3(1, (int)(M / 128)), 256, 0, stream>>>(
      xb_p, xpw_p, xdbl_p, nullptr, (int)M, 128, DI_, DI_, DI_, 128);

  // 5) dt projection: dt(4096,2048) = softplus(dt_in(4096,64) x dt_w(2048,64)^T + dt_b)
  slice_cvt_dtin<<<(int)(M * DR_ / 256), 256, 0, stream>>>(xdbl_p, dtin_p);
  gemm_bf16_wmma<EPI_SOFTPLUS_BIAS><<<dim3(DI_ / 128, (int)(M / 128)), 256, 0, stream>>>(
      dtin_p, dtw_p, dtf_p, dtb, (int)M, DI_, DR_, DR_, DR_, DI_);

  // 6) recurrent selective scan
  scan_kernel<<<dim3(DI_ / 16, B_), 256, 0, stream>>>(dtf_p, xdbl_p, xf_p, Alog, yf_p);

  // 7) gating
  gate_kernel<<<(int)(M * DI_ / 256), 256, 0, stream>>>(yf_p, xf_p, xz_p, Dp, yg_p);

  // 8) out_proj: out(4096,1024) = yg(4096,2048) x out_proj_w(1024,2048)^T
  gemm_bf16_wmma<EPI_NONE><<<dim3(DM_ / 128, (int)(M / 128)), 256, 0, stream>>>(
      yg_p, opw_p, (float*)d_out, nullptr, (int)M, DM_, DI_, DI_, DI_, DM_);
}
